// MultiHeadAttention_39926015983691
// MI455X (gfx1250) — compile-verified, matrix operand order checked by reading, not run on hardware
//
#include <hip/hip_runtime.h>
#include <hip/hip_bf16.h>

// ---------------------------------------------------------------------------
// MHA forward for MI455X (gfx1250, wave32, WMMA 16x16x32 f16)
//   B=4, S=2048, D_MODEL=1024, H=16, DK=64
// Pipeline:
//   1) Qh = q@Wq^T+bq  -> f16 [B,H,S,64]
//      Kh = k@Wk^T+bk  -> f16 [B,H,S,64]
//      VhT= v@Wv^T+bv  -> f16 [B,H,64,S]   (transposed for PV B-fragments)
//   2) flash attention -> f16 AO [B,S,1024] (head-concat)
//   3) out = AO@Wo^T+bo -> f32 d_out [B,S,1024]
// ---------------------------------------------------------------------------

#define DM   1024
#define SEQ  2048
#define NB   4
#define NH   16
#define DK   64

typedef _Float16 v16h __attribute__((ext_vector_type(16)));
typedef _Float16 v8h  __attribute__((ext_vector_type(8)));
typedef float    v8f  __attribute__((ext_vector_type(8)));
typedef int      v4i  __attribute__((ext_vector_type(4)));

typedef __attribute__((address_space(1))) v4i* gv4i_ptr;
typedef __attribute__((address_space(3))) v4i* lv4i_ptr;

#if __has_builtin(__builtin_amdgcn_global_load_async_to_lds_b128)
#define HAVE_ASYNC_LDS 1
#else
#define HAVE_ASYNC_LDS 0
#endif

static __device__ __forceinline__ v16h mk16(v8h lo, v8h hi) {
  return __builtin_shufflevector(lo, hi, 0,1,2,3,4,5,6,7,8,9,10,11,12,13,14,15);
}

static __device__ __forceinline__ v8f wmma_f16(v16h a, v16h b, v8f c) {
  // (neg_a, A, neg_b, B, c_mod, C, reuse_a, reuse_b)
  return __builtin_amdgcn_wmma_f32_16x16x32_f16(false, a, false, b, (short)0, c,
                                                false, false);
}

static __device__ __forceinline__ v8h cvt8(float4 a, float4 b) {
  v8h r;
  r[0] = (_Float16)a.x; r[1] = (_Float16)a.y; r[2] = (_Float16)a.z; r[3] = (_Float16)a.w;
  r[4] = (_Float16)b.x; r[5] = (_Float16)b.y; r[6] = (_Float16)b.z; r[7] = (_Float16)b.w;
  return r;
}

// ---------------------------------------------------------------------------
// GEMM: C[M=8192, N=1024] = X[M,1024] @ W[N,1024]^T + bias[N]
// XMODE: 0 = X is f32 (convert to f16 on LDS stage), 1 = X is f16
// OMODE: 0 = f32 plain [M,N]
//        1 = f16 head-split      [B,H,S,DK]   (n = h*64+d, m = b*S+s)
//        2 = f16 head-split-T    [B,H,DK,S]
// Block: 256 threads (8 waves), tile 128(M) x 128(N), K-step 32.
// Wave tile: 32(M) x 64(N) -> 2x4 WMMA accumulators, 8 wmma per K-step.
// ---------------------------------------------------------------------------
template <int XMODE, int OMODE>
__global__ __launch_bounds__(256)
void gemm_xwT(const void* __restrict__ Xv, const float* __restrict__ W,
              const float* __restrict__ bias, void* __restrict__ outv) {
  __shared__ __align__(16) _Float16 Xs[128][40];  // 32 data + 8 pad
  __shared__ __align__(16) _Float16 Ws[128][40];

  const int tid  = threadIdx.x;
  const int lane = tid & 31;
  const int w    = tid >> 5;
  const int mBase = blockIdx.x * 128;
  const int nBase = blockIdx.y * 128;
  const int m0 = (w >> 1) * 32;
  const int n0 = (w & 1) * 64;

  // fragment lane coords (16-bit A/B layout)
  const int fr   = lane & 15;
  const int fk   = (lane >> 4) * 8;
  const int hi16 = lane >> 4;
  const int col  = lane & 15;

  // cooperative tile-load coords: 128 rows x 32 cols, 16 elements per thread
  const int xr = tid >> 1, xc = (tid & 1) * 16;

  v8f acc[2][4] = {};

  for (int kb = 0; kb < DM; kb += 32) {
    if (XMODE == 0) {
      const float* X = (const float*)Xv;
      const float4* src = (const float4*)&X[(size_t)(mBase + xr) * DM + kb + xc];
      float4 a0 = src[0], a1 = src[1], a2 = src[2], a3 = src[3];
      *(v8h*)&Xs[xr][xc]     = cvt8(a0, a1);
      *(v8h*)&Xs[xr][xc + 8] = cvt8(a2, a3);
      __builtin_prefetch(&X[(size_t)(mBase + xr) * DM + kb + 32 + xc], 0, 1);
    } else {
      const _Float16* X = (const _Float16*)Xv;
      *(v8h*)&Xs[xr][xc]     = *(const v8h*)&X[(size_t)(mBase + xr) * DM + kb + xc];
      *(v8h*)&Xs[xr][xc + 8] = *(const v8h*)&X[(size_t)(mBase + xr) * DM + kb + xc + 8];
      __builtin_prefetch(&X[(size_t)(mBase + xr) * DM + kb + 32 + xc], 0, 1);
    }
    {
      const float4* src = (const float4*)&W[(size_t)(nBase + xr) * DM + kb + xc];
      float4 b0 = src[0], b1 = src[1], b2 = src[2], b3 = src[3];
      *(v8h*)&Ws[xr][xc]     = cvt8(b0, b1);
      *(v8h*)&Ws[xr][xc + 8] = cvt8(b2, b3);
      __builtin_prefetch(&W[(size_t)(nBase + xr) * DM + kb + 32 + xc], 0, 1);
    }
    __syncthreads();

    v16h a[2], b[4];
#pragma unroll
    for (int i = 0; i < 2; i++)
      a[i] = mk16(*(const v8h*)&Xs[m0 + i * 16 + fr][fk],
                  *(const v8h*)&Xs[m0 + i * 16 + fr][fk + 16]);
#pragma unroll
    for (int j = 0; j < 4; j++)
      b[j] = mk16(*(const v8h*)&Ws[n0 + j * 16 + fr][fk],
                  *(const v8h*)&Ws[n0 + j * 16 + fr][fk + 16]);
#pragma unroll
    for (int i = 0; i < 2; i++)
#pragma unroll
      for (int j = 0; j < 4; j++)
        acc[i][j] = wmma_f16(a[i], b[j], acc[i][j]);
    __syncthreads();
  }

  // store: C layout = VGPR v -> row (v + 8*hi16), lane col
#pragma unroll
  for (int j = 0; j < 4; j++) {
    const int n = nBase + n0 + j * 16 + col;
    const float bv = bias[n];
#pragma unroll
    for (int i = 0; i < 2; i++) {
#pragma unroll
      for (int v = 0; v < 8; v++) {
        const int m = mBase + m0 + i * 16 + v + 8 * hi16;
        const float val = acc[i][j][v] + bv;
        if (OMODE == 0) {
          ((float*)outv)[(size_t)m * DM + n] = val;
        } else {
          const int bi = m >> 11, s = m & (SEQ - 1);
          const int hh = n >> 6,  d = n & (DK - 1);
          size_t o;
          if (OMODE == 1)
            o = ((((size_t)bi * NH + hh) * SEQ + s) * DK + d);
          else
            o = ((((size_t)bi * NH + hh) * DK + d) * SEQ + s);
          ((_Float16*)outv)[o] = (_Float16)val;
        }
      }
    }
  }
}

// ---------------------------------------------------------------------------
// Flash attention: one (b,h) per blockIdx.y, 128 query rows per block,
// 8 waves x 16 query rows each. Key blocks of 32, online softmax.
// Qh/Kh: [B,H,S,64] f16. VhT: [B,H,64,S] f16. AO: [B,S,1024] f16.
// K/V tiles staged with GLOBAL_LOAD_ASYNC_TO_LDS_B128 (ASYNCcnt) when the
// toolchain exposes the builtin; register-staged copy otherwise.
// ---------------------------------------------------------------------------
__global__ __launch_bounds__(256)
void flash_attn(const _Float16* __restrict__ Qh, const _Float16* __restrict__ Kh,
                const _Float16* __restrict__ VhT, _Float16* __restrict__ AO) {
  __shared__ __align__(16) _Float16 Ks[32][72];      // 32 keys x 64 dk (+pad)
  __shared__ __align__(16) _Float16 Vts[64][40];     // 64 dk x 32 keys (+pad)
  __shared__ __align__(16) _Float16 Ps[8][16][40];   // per-wave P tile 16x32

  const int tid  = threadIdx.x;
  const int lane = tid & 31;
  const int w    = tid >> 5;
  const int bh   = blockIdx.y;
  const int b    = bh >> 4;
  const int h    = bh & 15;
  const int qRow0 = blockIdx.x * 128 + w * 16;

  const size_t headBase = (size_t)bh * SEQ * DK;
  const int fr   = lane & 15;
  const int fk   = (lane >> 4) * 8;
  const int hi16 = lane >> 4;
  const int col  = lane & 15;

  // Q fragments (16 rows x 64 dk = two 16x32 A-frags), pre-scaled by 1/sqrt(64)
  v16h aq[2];
  {
    const _Float16* Qrow = Qh + headBase + (size_t)(qRow0 + fr) * DK;
#pragma unroll
    for (int i = 0; i < 2; i++) {
      v16h f = mk16(*(const v8h*)&Qrow[i * 32 + fk],
                    *(const v8h*)&Qrow[i * 32 + fk + 16]);
#pragma unroll
      for (int e = 0; e < 16; e++) f[e] = f[e] * (_Float16)0.125f;
      aq[i] = f;
    }
  }

  float mi[8], li[8];
  v8f oacc[4] = {};
#pragma unroll
  for (int v = 0; v < 8; v++) { mi[v] = -3.0e38f; li[v] = 0.0f; }

  const int kr = tid >> 3, kc = (tid & 7) * 8;  // Ks coop-load coords
  const int vr = tid >> 2, vc = (tid & 3) * 8;  // Vts coop-load coords
  const size_t vtBase = (size_t)bh * DK * SEQ;

  for (int kb = 0; kb < SEQ; kb += 32) {
    const _Float16* kSrc = &Kh[headBase + (size_t)(kb + kr) * DK + kc];
    const _Float16* vSrc = &VhT[vtBase + (size_t)vr * SEQ + kb + vc];
#if HAVE_ASYNC_LDS
    __builtin_amdgcn_global_load_async_to_lds_b128(
        (gv4i_ptr)kSrc, (lv4i_ptr)&Ks[kr][kc], 0, 0);
    __builtin_amdgcn_global_load_async_to_lds_b128(
        (gv4i_ptr)vSrc, (lv4i_ptr)&Vts[vr][vc], 0, 0);
    __builtin_amdgcn_s_wait_asynccnt(0);
#else
    *(v8h*)&Ks[kr][kc]  = *(const v8h*)kSrc;
    *(v8h*)&Vts[vr][vc] = *(const v8h*)vSrc;
#endif
    __builtin_prefetch(kSrc + 32 * DK, 0, 1);   // next key block of K
    __builtin_prefetch(vSrc + 32, 0, 1);        // next key block of V^T
    __syncthreads();

    // S = Q K^T : two 16x16 tiles over 32 keys
    v8f s0 = {}, s1 = {};
    {
      v16h bk;
      bk = mk16(*(const v8h*)&Ks[fr][fk],           *(const v8h*)&Ks[fr][fk + 16]);
      s0 = wmma_f16(aq[0], bk, s0);
      bk = mk16(*(const v8h*)&Ks[fr][32 + fk],      *(const v8h*)&Ks[fr][32 + fk + 16]);
      s0 = wmma_f16(aq[1], bk, s0);
      bk = mk16(*(const v8h*)&Ks[16 + fr][fk],      *(const v8h*)&Ks[16 + fr][fk + 16]);
      s1 = wmma_f16(aq[0], bk, s1);
      bk = mk16(*(const v8h*)&Ks[16 + fr][32 + fk], *(const v8h*)&Ks[16 + fr][32 + fk + 16]);
      s1 = wmma_f16(aq[1], bk, s1);
    }

    // online softmax update; row r = v + 8*hi16 lives across 16-lane groups,
    // xor masks 1/2/4/8 stay within a group.
#pragma unroll
    for (int v = 0; v < 8; v++) {
      float mx = fmaxf(s0[v], s1[v]);
      mx = fmaxf(mx, __shfl_xor(mx, 1));
      mx = fmaxf(mx, __shfl_xor(mx, 2));
      mx = fmaxf(mx, __shfl_xor(mx, 4));
      mx = fmaxf(mx, __shfl_xor(mx, 8));
      const float mnew  = fmaxf(mi[v], mx);
      const float alpha = __expf(mi[v] - mnew);
      const float p0 = __expf(s0[v] - mnew);
      const float p1 = __expf(s1[v] - mnew);
      float rs = p0 + p1;
      rs += __shfl_xor(rs, 1);
      rs += __shfl_xor(rs, 2);
      rs += __shfl_xor(rs, 4);
      rs += __shfl_xor(rs, 8);
      li[v] = li[v] * alpha + rs;
      mi[v] = mnew;
#pragma unroll
      for (int t = 0; t < 4; t++) oacc[t][v] *= alpha;
      Ps[w][v + 8 * hi16][col]      = (_Float16)p0;
      Ps[w][v + 8 * hi16][16 + col] = (_Float16)p1;
    }
    __syncthreads();

    // O += P @ V  (contraction over 32 keys, 4 dk tiles of 16)
    {
      const v16h ap = mk16(*(const v8h*)&Ps[w][fr][fk],
                           *(const v8h*)&Ps[w][fr][fk + 16]);
#pragma unroll
      for (int t = 0; t < 4; t++) {
        v16h bv = mk16(*(const v8h*)&Vts[t * 16 + fr][fk],
                       *(const v8h*)&Vts[t * 16 + fr][fk + 16]);
        oacc[t] = wmma_f16(ap, bv, oacc[t]);
      }
    }
    __syncthreads();
  }

  // normalize and store head-concat f16 [B,S,1024]
#pragma unroll
  for (int t = 0; t < 4; t++) {
    const int n = h * DK + t * 16 + col;
#pragma unroll
    for (int v = 0; v < 8; v++) {
      const int srow = qRow0 + v + 8 * hi16;
      AO[((size_t)b * SEQ + srow) * DM + n] = (_Float16)(oacc[t][v] / li[v]);
    }
  }
}

// ---------------------------------------------------------------------------
extern "C" void kernel_launch(void* const* d_in, const int* in_sizes, int n_in,
                              void* d_out, int out_size, void* d_ws, size_t ws_size,
                              hipStream_t stream) {
  (void)in_sizes; (void)n_in; (void)out_size; (void)ws_size;
  const float* q  = (const float*)d_in[0];
  const float* k  = (const float*)d_in[1];
  const float* v  = (const float*)d_in[2];
  const float* Wq = (const float*)d_in[3];
  const float* bq = (const float*)d_in[4];
  const float* Wk = (const float*)d_in[5];
  const float* bk = (const float*)d_in[6];
  const float* Wv = (const float*)d_in[7];
  const float* bv = (const float*)d_in[8];
  const float* Wo = (const float*)d_in[9];
  const float* bo = (const float*)d_in[10];

  const size_t HSZ = (size_t)NB * NH * SEQ * DK;  // 8,388,608 elems
  _Float16* Qh  = (_Float16*)d_ws;
  _Float16* Kh  = Qh + HSZ;
  _Float16* VhT = Kh + HSZ;
  _Float16* AO  = VhT + HSZ;                       // [B,S,1024] f16

  const dim3 gblk(256);
  const dim3 ggrid(64, 8);    // 8192/128 x 1024/128
  gemm_xwT<0, 1><<<ggrid, gblk, 0, stream>>>(q, Wq, bq, Qh);
  gemm_xwT<0, 1><<<ggrid, gblk, 0, stream>>>(k, Wk, bk, Kh);
  gemm_xwT<0, 2><<<ggrid, gblk, 0, stream>>>(v, Wv, bv, VhT);

  flash_attn<<<dim3(SEQ / 128, NB * NH), gblk, 0, stream>>>(Qh, Kh, VhT, AO);

  gemm_xwT<1, 0><<<ggrid, gblk, 0, stream>>>(AO, Wo, bo, (float*)d_out);
}
